// REN_40175124087397
// MI455X (gfx1250) — matrix operand order repaired and checked by reading, not compiled, not run on hardware
//
#include <hip/hip_runtime.h>

#define NX     64
#define NU     64
#define NUNITS 256
#define NY     32
#define NZ     384
#define BATCHN 32768

typedef __attribute__((ext_vector_type(16))) __bf16 bf16x16;
typedef __attribute__((ext_vector_type(8)))  __bf16 bf16x8;
typedef __attribute__((ext_vector_type(8)))  float  f32x8;

#if __has_builtin(__builtin_amdgcn_tanhf)
#define TANHF(x) __builtin_amdgcn_tanhf(x)
#else
#define TANHF(x) tanhf(x)
#endif

#define WMMA_BF16(A, B, C) \
  __builtin_amdgcn_wmma_f32_16x16x32_bf16(false, (A), false, (B), (short)0, (C), false, false)

__device__ __forceinline__ float bperm_f(int sidx, float v) {
  return __builtin_bit_cast(float,
      __builtin_amdgcn_ds_bpermute(sidx, __builtin_bit_cast(int, v)));
}

// ---------------------------------------------------------------------------
// Preprocessing: H(i,j) computed on demand.
// H = X^T X + 1e-3 I + T1 * lT1^T - T2 * lT2^T, with T1 = lT1*Rinv, T2 = lT2*Q
// ---------------------------------------------------------------------------
__device__ __forceinline__ float hval(int i, int j,
    const float* __restrict__ X,  const float* __restrict__ lT1,
    const float* __restrict__ T1, const float* __restrict__ lT2,
    const float* __restrict__ T2) {
  float s = (i == j) ? 0.001f : 0.0f;
  for (int t = 0; t < NZ; ++t) s += X[t * NZ + i] * X[t * NZ + j];
  for (int a = 0; a < NU; ++a) s += T1[i * NU + a] * lT1[j * NU + a];
  for (int a = 0; a < NY; ++a) s -= T2[i * NY + a] * lT2[j * NY + a];
  return s;
}

// lT1 (384x64) and lT2 (384x32).  grid=(384), block=(64): i=blockIdx, a=threadIdx
__global__ void prep_lt(const float* __restrict__ B_2, const float* __restrict__ C_2,
                        const float* __restrict__ D_12, const float* __restrict__ D_21,
                        const float* __restrict__ S_t,
                        float* __restrict__ lT1, float* __restrict__ lT2) {
  int i = blockIdx.x, a = threadIdx.x;
  float v;
  if (i < NX) {                       // C2_^T rows: C2_ = S_t @ C_2  (64x64)
    v = 0.f;
    for (int t = 0; t < NY; ++t) v += S_t[a * NY + t] * C_2[t * NX + i];
  } else if (i < NX + NUNITS) {       // D21_^T rows: D21_ = S_t@D_21 - D_12^T
    int un = i - NX;
    v = 0.f;
    for (int t = 0; t < NY; ++t) v += S_t[a * NY + t] * D_21[t * NUNITS + un];
    v -= D_12[un * NU + a];
  } else {                            // B_2 rows
    v = B_2[(i - NX - NUNITS) * NU + a];
  }
  lT1[i * NU + a] = v;
  if (a < NY) {
    float v2;
    if (i < NX)               v2 = C_2[a * NX + i];
    else if (i < NX + NUNITS) v2 = D_21[a * NUNITS + (i - NX)];
    else                      v2 = 0.f;
    lT2[i * NY + a] = v2;
  }
}

// T1 = lT1 @ Rinv (384x64), T2 = lT2 @ Q (384x32). grid=(384), block=(64)
__global__ void prep_t12(const float* __restrict__ lT1, const float* __restrict__ lT2,
                         const float* __restrict__ Rinv, const float* __restrict__ Q,
                         float* __restrict__ T1, float* __restrict__ T2) {
  int i = blockIdx.x, a = threadIdx.x;
  float s = 0.f;
  for (int bb = 0; bb < NU; ++bb) s += lT1[i * NU + bb] * Rinv[bb * NU + a];
  T1[i * NU + a] = s;
  if (a < NY) {
    float s2 = 0.f;
    for (int bb = 0; bb < NY; ++bb) s2 += lT2[i * NY + bb] * Q[bb * NY + a];
    T2[i * NY + a] = s2;
  }
}

// Wt[k][n] = Drows[n][k] = -H22[n-1][k] for n>=1 && k<=n-2, else 0  (256x256 bf16)
// grid=(256) n, block=(256) k
__global__ void prep_wt(const float* __restrict__ X,  const float* __restrict__ lT1,
                        const float* __restrict__ T1, const float* __restrict__ lT2,
                        const float* __restrict__ T2, unsigned short* __restrict__ wt_us) {
  int n = blockIdx.x, k = threadIdx.x;
  float v = 0.f;
  if (n >= 1 && k <= n - 2)
    v = -hval(NX + n - 1, NX + k, X, lT1, T1, lT2, T2);
  ((__bf16*)wt_us)[k * NUNITS + n] = (__bf16)v;
}

// CatB (128x256 bf16): rows 0..63 = C_1^T = -H21^T, rows 64..127 = D_12^T.
// Also invlam[n] = 2 / H22[n][n].  grid=(256) n, block=(128) k
__global__ void prep_catb(const float* __restrict__ X,  const float* __restrict__ lT1,
                          const float* __restrict__ T1, const float* __restrict__ lT2,
                          const float* __restrict__ T2, const float* __restrict__ D_12,
                          unsigned short* __restrict__ catb_us,
                          float* __restrict__ invlam) {
  int n = blockIdx.x, k = threadIdx.x;
  float v;
  if (k < NX) v = -hval(NX + n, k, X, lT1, T1, lT2, T2);   // C_1[n][k]
  else        v = D_12[n * NU + (k - NX)];                  // D_12[n][k-64]
  ((__bf16*)catb_us)[k * NUNITS + n] = (__bf16)v;
  if (k == 0)
    invlam[n] = 2.0f / hval(NX + n, NX + n, X, lT1, T1, lT2, T2);
}

// D21t (256x32 bf16) and C2t (64x32 bf16): transposed copies. grid=(320), block=(32)
__global__ void prep_misc(const float* __restrict__ C_2, const float* __restrict__ D_21,
                          unsigned short* __restrict__ d21t_us,
                          unsigned short* __restrict__ c2t_us) {
  int row = blockIdx.x, n = threadIdx.x;
  if (row < NUNITS) ((__bf16*)d21t_us)[row * NY + n] = (__bf16)D_21[n * NUNITS + row];
  else { int k = row - NUNITS; ((__bf16*)c2t_us)[k * NY + n] = (__bf16)C_2[n * NX + k]; }
}

// ---------------------------------------------------------------------------
// Main: one wave handles a 16-row batch tile. 2 waves / 64-thread block.
// ---------------------------------------------------------------------------
// A-fragment (bf16 16x32): lane m = lane&15, half = lane>>4;
//   elems 0..7  -> K = 8*half + e, elems 8..15 -> K = 16 + 8*half + (e-8) = e + 8 + 8*half
__device__ __forceinline__ bf16x16 w_pair_frag(const __bf16* wrow, int kbase, int half) {
  const __bf16* p = wrow + kbase + 8 * half;
  bf16x8 lo = *(const bf16x8*)p;
  bf16x8 hi = *(const bf16x8*)(p + 16);
  bf16x16 a;
#pragma unroll
  for (int e = 0; e < 8; ++e) { a[e] = lo[e]; a[e + 8] = hi[e]; }
  return a;
}

__global__ __launch_bounds__(64) void ren_main(
    const float* __restrict__ u, const float* __restrict__ x0,
    const float* __restrict__ bvec,
    const unsigned short* __restrict__ catb_us, const unsigned short* __restrict__ wt_us,
    const unsigned short* __restrict__ d21t_us, const unsigned short* __restrict__ c2t_us,
    const float* __restrict__ invlam, float* __restrict__ out) {
  const __bf16* CatB = (const __bf16*)catb_us;
  const __bf16* Wt   = (const __bf16*)wt_us;
  const __bf16* D21t = (const __bf16*)d21t_us;
  const __bf16* C2t  = (const __bf16*)c2t_us;

  __shared__ float  ldsV[2][16 * NUNITS];   // v tile, fp32
  __shared__ __bf16 ldsW[2][16 * NUNITS];   // w tile, bf16 (A-operand source)

  const int lane = threadIdx.x & 31;
  const int wid  = threadIdx.x >> 5;
  const int l15  = lane & 15;
  const int half = lane >> 4;
  const int m0   = (blockIdx.x * 2 + wid) * 16;

  // ---- Load x||u A-fragments (K = 128 over [x2 (64) | u2 (64)]), fp32 -> bf16
  bf16x16 Axu[4];
  {
    const float* xrow = x0 + (size_t)(m0 + l15) * NX;
    const float* urow = u  + (size_t)(m0 + l15) * NU;
#pragma unroll
    for (int kb = 0; kb < 4; ++kb) {
      const float* src = (kb < 2) ? xrow : urow;
      const int kbase = (kb & 1) * 32;
      bf16x16 a;
#pragma unroll
      for (int e = 0; e < 8; ++e) {
        a[e]     = (__bf16)src[kbase + 8 * half + e];
        a[e + 8] = (__bf16)src[kbase + 16 + 8 * half + e];
      }
      Axu[kb] = a;
    }
  }

  // ---- v = [x2|u2] @ CatB + b_w  -> LDS (16 tiles of N=16, K=128 in 4 WMMAs)
#pragma unroll
  for (int tile = 0; tile < 16; ++tile) {
    const float bias = bvec[NX + tile * 16 + l15];
    f32x8 acc;
#pragma unroll
    for (int r = 0; r < 8; ++r) acc[r] = bias;
#pragma unroll
    for (int kb = 0; kb < 4; ++kb) {
      bf16x16 B;
      for (int s = 0; s < 16; ++s) B[s] = CatB[(kb * 32 + 8 * half + s + (s & 8)) * NUNITS + tile * 16 + l15];
      acc = WMMA_BF16(Axu[kb], B, acc);
    }
#pragma unroll
    for (int r = 0; r < 8; ++r)
      ldsV[wid][(r + 8 * half) * NUNITS + tile * 16 + l15] = acc[r];
  }

  // ---- Blocked triangular scan over 16 unit-blocks
  const __bf16* wrow = &ldsW[wid][l15 * NUNITS];
#pragma unroll 1
  for (int b = 0; b < 16; ++b) {
    f32x8 acc;
#pragma unroll
    for (int r = 0; r < 8; ++r)
      acc[r] = ldsV[wid][(r + 8 * half) * NUNITS + b * 16 + l15];

    // cross-block: v[:,b] += w[:,0:16b] @ Wt[0:16b, b-block]
    const int npair = b >> 1;
#pragma unroll 1
    for (int kp = 0; kp < npair; ++kp) {
      bf16x16 Aw = w_pair_frag(wrow, kp * 32, half);
      bf16x16 Bw;
      for (int s = 0; s < 16; ++s) Bw[s] = Wt[(kp * 32 + 8 * half + s + (s & 8)) * NUNITS + b * 16 + l15];
      acc = WMMA_BF16(Aw, Bw, acc);
    }
    if (b & 1) {  // leftover single previous block: zero upper K-half of A
      const int p = b - 1;
      const __bf16* pp = wrow + p * 16 + 8 * half;
      bf16x8 lo = *(const bf16x8*)pp;
      bf16x16 Aw;
#pragma unroll
      for (int e = 0; e < 8; ++e) { Aw[e] = lo[e]; Aw[e + 8] = (__bf16)0.0f; }
      bf16x16 Bw;
      for (int s = 0; s < 16; ++s) Bw[s] = Wt[(p * 16 + 8 * half + s + (s & 8)) * NUNITS + b * 16 + l15];
      acc = WMMA_BF16(Aw, Bw, acc);
    }

    // within-block: 16 sequential tanh steps, lane-broadcast via ds_bpermute
    const float llam = invlam[b * 16 + l15];
#pragma unroll
    for (int t = 0; t < 16; ++t) {
      const int sidx = (((lane >> 4) << 4) + t) << 2;  // src lane = t + 16*half
      const float il = bperm_f(sidx, llam);
      float wr[8];
#pragma unroll
      for (int r = 0; r < 8; ++r)
        wr[r] = TANHF(bperm_f(sidx, acc[r]) * il);
      const float wc = (float)Wt[(b * 16 + t) * NUNITS + b * 16 + l15];
#pragma unroll
      for (int r = 0; r < 8; ++r) acc[r] += wr[r] * wc;
      if (l15 == t) {
#pragma unroll
        for (int r = 0; r < 8; ++r)
          ldsW[wid][(r + 8 * half) * NUNITS + b * 16 + t] = (__bf16)wr[r];
      }
    }
  }

  // ---- y = x2 @ C_2^T + w @ D_21^T + b_y   (2 tiles of N=16)
#pragma unroll
  for (int tile = 0; tile < 2; ++tile) {
    const float by = bvec[NX + NUNITS + tile * 16 + l15];
    f32x8 yacc;
#pragma unroll
    for (int r = 0; r < 8; ++r) yacc[r] = by;
#pragma unroll
    for (int kb = 0; kb < 2; ++kb) {
      bf16x16 B;
      for (int s = 0; s < 16; ++s) B[s] = C2t[(kb * 32 + 8 * half + s + (s & 8)) * NY + tile * 16 + l15];
      yacc = WMMA_BF16(Axu[kb], B, yacc);
    }
#pragma unroll 1
    for (int kp = 0; kp < 8; ++kp) {
      bf16x16 Aw = w_pair_frag(wrow, kp * 32, half);
      bf16x16 Bw;
      for (int s = 0; s < 16; ++s) Bw[s] = D21t[(kp * 32 + 8 * half + s + (s & 8)) * NY + tile * 16 + l15];
      yacc = WMMA_BF16(Aw, Bw, yacc);
    }
#pragma unroll
    for (int r = 0; r < 8; ++r)
      out[(size_t)(m0 + r + 8 * half) * NY + tile * 16 + l15] = yacc[r];
  }
}

// ---------------------------------------------------------------------------
extern "C" void kernel_launch(void* const* d_in, const int* in_sizes, int n_in,
                              void* d_out, int out_size, void* d_ws, size_t ws_size,
                              hipStream_t stream) {
  (void)in_sizes; (void)n_in; (void)out_size; (void)ws_size;
  const float* u    = (const float*)d_in[0];
  const float* x0   = (const float*)d_in[1];
  const float* B_2  = (const float*)d_in[2];
  const float* C_2  = (const float*)d_in[3];
  const float* D_12 = (const float*)d_in[4];
  const float* D_21 = (const float*)d_in[5];
  const float* bvec = (const float*)d_in[6];
  const float* X    = (const float*)d_in[7];
  const float* S_t  = (const float*)d_in[9];
  const float* Q    = (const float*)d_in[10];
  const float* Rinv = (const float*)d_in[11];

  char* ws = (char*)d_ws;
  float*          lT1    = (float*)(ws + 0);        // 384*64*4  = 98304
  float*          lT2    = (float*)(ws + 98304);    // 384*32*4  = 49152
  float*          T1     = (float*)(ws + 147456);   // 98304
  float*          T2     = (float*)(ws + 245760);   // 49152
  float*          invlam = (float*)(ws + 294912);   // 1024
  unsigned short* CatB   = (unsigned short*)(ws + 295936);  // 128*256*2 = 65536
  unsigned short* Wt     = (unsigned short*)(ws + 361472);  // 256*256*2 = 131072
  unsigned short* D21t   = (unsigned short*)(ws + 492544);  // 256*32*2  = 16384
  unsigned short* C2t    = (unsigned short*)(ws + 508928);  // 64*32*2   = 4096

  prep_lt  <<<NZ, NU, 0, stream>>>(B_2, C_2, D_12, D_21, S_t, lT1, lT2);
  prep_t12 <<<NZ, NU, 0, stream>>>(lT1, lT2, Rinv, Q, T1, T2);
  prep_wt  <<<NUNITS, NUNITS, 0, stream>>>(X, lT1, T1, lT2, T2, Wt);
  prep_catb<<<NUNITS, 128, 0, stream>>>(X, lT1, T1, lT2, T2, D_12, CatB, invlam);
  prep_misc<<<NUNITS + NX, NY, 0, stream>>>(C_2, D_21, D21t, C2t);

  ren_main<<<BATCHN / 32, 64, 0, stream>>>(u, x0, bvec, CatB, Wt, D21t, C2t, invlam,
                                           (float*)d_out);
}
